// GraphEncoder_83837761618269
// MI455X (gfx1250) — matrix operand order repaired and checked by reading, not compiled, not run on hardware
//
#include <hip/hip_runtime.h>
#include <hip/hip_bf16.h>

// ---------------- constants (match reference) ----------------
#define NN 100000
#define EE 600000
#define VOCAB 51
#define HID 128
#define HEADS 2
#define NG 256
#define DGAT 256   // HEADS*HID

#define AS1 __attribute__((address_space(1)))
#define AS3 __attribute__((address_space(3)))

#if __has_builtin(__builtin_amdgcn_global_load_async_to_lds_b128)
#define HAVE_ASYNC_LDS 1
#else
#define HAVE_ASYNC_LDS 0
#endif

// ---------------- WMMA / async vector types ----------------
typedef __attribute__((ext_vector_type(16))) __bf16 v16bf;
typedef __attribute__((ext_vector_type(8)))  float  v8f;
typedef int v4i __attribute__((vector_size(16)));   // matches async-LDS builtin param

__device__ __forceinline__ __bf16 us_to_bf(unsigned short u) {
    union { unsigned short u; __bf16 b; } c; c.u = u; return c.b;
}
__device__ __forceinline__ unsigned short f32_to_bf16_rne(float f) {
    unsigned int u = __float_as_uint(f);
    return (unsigned short)((u + 0x7FFFu + ((u >> 16) & 1u)) >> 16);
}

// ---------------- K0a: zero a float region ----------------
__global__ void kzero(float* __restrict__ p, int n) {
    int i = blockIdx.x * 256 + threadIdx.x;
    if (i < n) p[i] = 0.0f;
}

// ---------------- K0b: deg=1 (self loop), count nodes per graph ----------------
__global__ void k0b_init(int* __restrict__ deg, float* __restrict__ cnt,
                         const int* __restrict__ batch) {
    int i = blockIdx.x * 256 + threadIdx.x;
    if (i < NN) {
        deg[i] = 1;
        atomicAdd(&cnt[batch[i]], 1.0f);
    }
}

// ---------------- K1a: table51 = emb @ W  (64x256, rows>=51 zero) ----------------
__global__ void k1a_table(const float* __restrict__ emb, const float* __restrict__ W,
                          float* __restrict__ tab) {
    int idx = blockIdx.x * 256 + threadIdx.x;   // 64*256 threads
    int v = idx >> 8, c = idx & 255;
    float s = 0.0f;
    if (v < VOCAB) {
        #pragma unroll 4
        for (int k = 0; k < HID; ++k)
            s = fmaf(emb[v * HID + k], W[k * DGAT + c], s);
    }
    tab[idx] = s;
}

// ---------------- K1b: per-vocab attention scalars ----------------
__global__ void k1b_att(const float* __restrict__ tab,
                        const float* __restrict__ att_s, const float* __restrict__ att_d,
                        float* __restrict__ asrc, float* __restrict__ adst) {
    int idx = threadIdx.x;          // 128 threads: v in [0,64), h in {0,1}
    int v = idx >> 1, h = idx & 1;
    float s = 0.0f, d = 0.0f;
    #pragma unroll 4
    for (int k = 0; k < HID; ++k) {
        float t = tab[v * DGAT + h * HID + k];
        s = fmaf(t, att_s[h * HID + k], s);
        d = fmaf(t, att_d[h * HID + k], d);
    }
    asrc[v * 2 + h] = s;
    adst[v * 2 + h] = d;
}

// ---------------- K1c: pre-pack W2 into per-lane WMMA B-fragment layout ----------------
// BPACK[((ct*8 + kk)*32 + lane)*16 + e] = bf16(W2[kB][cB])
//   kB = kk*32 + (e<8 ? 0 : 16) + (lane>=16 ? 8 : 0) + (e&7) ; cB = ct*16 + (lane&15)
__global__ void k1c_pack(const float* __restrict__ w2, unsigned short* __restrict__ bpack) {
    int idx = blockIdx.x * 256 + threadIdx.x;   // 8*8*32*16 = 32768
    if (idx < 32768) {
        int e    = idx & 15;
        int lane = (idx >> 4) & 31;
        int kk   = (idx >> 9) & 7;
        int ct   = idx >> 12;
        int kB = kk * 32 + ((e < 8) ? 0 : 16) + ((lane >= 16) ? 8 : 0) + (e & 7);
        int cB = ct * 16 + (lane & 15);
        bpack[idx] = f32_to_bf16_rne(w2[kB * HID + cB]);
    }
}

// ---------------- K2: count incoming edges ----------------
__global__ void k2_count(const int* __restrict__ dst, int* __restrict__ deg) {
    int i = blockIdx.x * 256 + threadIdx.x;
    if (i < EE) atomicAdd(&deg[dst[i]], 1);
}

// ---------------- K3: exclusive scan of deg -> offs, curs (single block) ----------------
__global__ void k3_scan(const int* __restrict__ deg, int* __restrict__ offs,
                        int* __restrict__ curs) {
    __shared__ int sd[1024];
    int tid = threadIdx.x;
    int carry = 0;
    for (int base = 0; base < NN; base += 1024) {
        int i = base + tid;
        int v = (i < NN) ? deg[i] : 0;
        sd[tid] = v;
        __syncthreads();
        for (int off = 1; off < 1024; off <<= 1) {
            int t = (tid >= off) ? sd[tid - off] : 0;
            __syncthreads();
            sd[tid] += t;
            __syncthreads();
        }
        int excl = sd[tid] - v;
        if (i < NN) { offs[i] = carry + excl; curs[i] = carry + excl; }
        __syncthreads();
        carry += sd[1023];
        __syncthreads();
    }
    if (tid == 0) offs[NN] = carry;   // == EE + NN
}

// ---------------- K4: scatter edges (and self loops) into CSR ----------------
__global__ void k4_scatter(const int* __restrict__ src, const int* __restrict__ dst,
                           int* __restrict__ curs, int* __restrict__ csrc) {
    int i = blockIdx.x * 256 + threadIdx.x;
    if (i < EE + NN) {
        int d, s;
        if (i < EE) { d = dst[i]; s = src[i]; }
        else        { d = i - EE; s = d; }
        int pos = atomicAdd(&curs[d], 1);
        csrc[pos] = s;
    }
}

// ---------------- K5: per-node GAT softmax-aggregate + LayerNorm -> bf16 hn ----------------
__global__ void k5_gat(const int* __restrict__ xv, const int* __restrict__ offs,
                       const int* __restrict__ csrc, const float* __restrict__ tab51,
                       const float* __restrict__ asrc, const float* __restrict__ adst,
                       const float* __restrict__ bias, const float* __restrict__ gamma,
                       const float* __restrict__ beta, unsigned short* __restrict__ hn) {
    int lane = threadIdx.x & 31;
    int wave = threadIdx.x >> 5;
    int n = blockIdx.x * 8 + wave;        // NN divisible by 8 -> no tail

    int v = xv[n];
    float ad0 = adst[v * 2 + 0], ad1 = adst[v * 2 + 1];
    int beg = offs[n], end = offs[n + 1];

    // pass 1: per-head max (wave-uniform loop; degree >= 1 via self loop)
    float m0 = -1e30f, m1 = -1e30f;
    for (int i = beg; i < end; ++i) {
        int sv = xv[csrc[i]];
        __builtin_prefetch(tab51 + (size_t)sv * DGAT + lane, 0, 1);  // global_prefetch_b8
        float e0 = asrc[sv * 2 + 0] + ad0; e0 = e0 > 0.f ? e0 : 0.2f * e0;
        float e1 = asrc[sv * 2 + 1] + ad1; e1 = e1 > 0.f ? e1 : 0.2f * e1;
        m0 = fmaxf(m0, e0); m1 = fmaxf(m1, e1);
    }

    // pass 2: accumulate exp-weighted rows + denominators
    float acc[8];
    #pragma unroll
    for (int j = 0; j < 8; ++j) acc[j] = 0.0f;
    float l0 = 0.0f, l1 = 0.0f;
    for (int i = beg; i < end; ++i) {
        int sv = xv[csrc[i]];
        float e0 = asrc[sv * 2 + 0] + ad0; e0 = e0 > 0.f ? e0 : 0.2f * e0;
        float e1 = asrc[sv * 2 + 1] + ad1; e1 = e1 > 0.f ? e1 : 0.2f * e1;
        float w0 = __expf(e0 - m0), w1 = __expf(e1 - m1);
        l0 += w0; l1 += w1;
        const float* row = tab51 + (size_t)sv * DGAT;   // 52KB table: L2-resident
        #pragma unroll
        for (int j = 0; j < 8; ++j) {
            float wj = (j < 4) ? w0 : w1;               // cols 0..127 head0, 128..255 head1
            acc[j] = fmaf(wj, row[lane + 32 * j], acc[j]);
        }
    }
    float inv0 = 1.0f / l0, inv1 = 1.0f / l1;

    // h = out + bias; LayerNorm over 256 via wave32 shuffles
    float h[8], psum = 0.0f;
    #pragma unroll
    for (int j = 0; j < 8; ++j) {
        int c = lane + 32 * j;
        h[j] = acc[j] * ((j < 4) ? inv0 : inv1) + bias[c];
        psum += h[j];
    }
    for (int o = 16; o > 0; o >>= 1) psum += __shfl_xor(psum, o, 32);
    float mu = psum * (1.0f / 256.0f);
    float vs = 0.0f;
    #pragma unroll
    for (int j = 0; j < 8; ++j) { float d = h[j] - mu; vs += d * d; }
    for (int o = 16; o > 0; o >>= 1) vs += __shfl_xor(vs, o, 32);
    float rs = rsqrtf(vs * (1.0f / 256.0f) + 1e-5f);

    unsigned short* dstp = hn + (size_t)n * DGAT;
    #pragma unroll
    for (int j = 0; j < 8; ++j) {
        int c = lane + 32 * j;
        float y = (h[j] - mu) * rs * gamma[c] + beta[c];
        dstp[c] = f32_to_bf16_rne(y);
    }
}

// ---------------- K6: FFN via v_wmma_f32_16x16x32_bf16, fused bias+lrelu+pool ----------------
__global__ void k6_ffn(const unsigned short* __restrict__ hn,
                       const unsigned short* __restrict__ bpack,
                       const float* __restrict__ b2,
                       const int* __restrict__ batch,
                       float* __restrict__ out_ffn,
                       float* __restrict__ pooled) {
    __shared__ unsigned short sA[16][DGAT];   // 16-node hn tile, bf16 bits (8KB)
    int tid = threadIdx.x;                    // 256 threads = 8 waves
    int base16 = blockIdx.x * 16;             // NN divisible by 16 -> no tail

    {   // stage hn tile: 32B per thread
        int r   = tid >> 4;
        int c16 = (tid & 15) << 4;
        const unsigned short* gsrc = hn + (size_t)(base16 + r) * DGAT + c16;
#if HAVE_ASYNC_LDS
        AS1 v4i* gp = (AS1 v4i*)(uintptr_t)gsrc;
        AS3 v4i* lp = (AS3 v4i*)(unsigned)(uintptr_t)&sA[r][c16];  // addr[31:0] == LDS offset
        __builtin_amdgcn_global_load_async_to_lds_b128(gp, lp, 0, 0);
        __builtin_amdgcn_global_load_async_to_lds_b128(gp, lp, 16, 0);
#if __has_builtin(__builtin_amdgcn_s_wait_asynccnt)
        __builtin_amdgcn_s_wait_asynccnt(0);
#else
        asm volatile("s_wait_asynccnt 0x0" ::: "memory");
#endif
#else
        const uint4* g4 = (const uint4*)gsrc;
        uint4* s4 = (uint4*)&sA[r][c16];
        s4[0] = g4[0];
        s4[1] = g4[1];
#endif
    }
    __syncthreads();

    int wave = tid >> 5, lane = tid & 31;
    int col0 = wave * 16;                     // this wave's 16 output columns
    int lhalf = lane & 15;
    bool hi = lane >= 16;

    // B fragments: pre-packed per-lane layout, 32B contiguous per (wave,kk,lane)
    const unsigned short* bbase = bpack + (((size_t)wave * 8) * 32 + lane) * 16;

    v8f acc = {0.f, 0.f, 0.f, 0.f, 0.f, 0.f, 0.f, 0.f};
    #pragma unroll
    for (int kk = 0; kk < 8; ++kk) {          // K = 256 in 8 steps of 32
        int kbase = kk * 32;
        v16bf a;
        // A (16x32 bf16): lanes 0-15 row=lane, K {0..7,16..23}; lanes 16-31 row=lane-16, K {8..15,24..31}
        int koff = hi ? 8 : 0;
        #pragma unroll
        for (int e = 0; e < 8; ++e) a[e]     = us_to_bf(sA[lhalf][kbase + koff + e]);
        #pragma unroll
        for (int e = 0; e < 8; ++e) a[8 + e] = us_to_bf(sA[lhalf][kbase + 16 + koff + e]);

        v16bf b = *(const v16bf*)(bbase + (size_t)kk * 32 * 16);   // 2x global_load_b128, L2-hot

        acc = __builtin_amdgcn_wmma_f32_16x16x32_bf16(false, a, false, b,
                                                      (short)0, acc, false, false);
    }

    // D layout: element r -> row (hi?8:0)+r, col = col0 + lhalf
    int c = col0 + lhalf;
    float bb = b2[c];
    int rbase = hi ? 8 : 0;
    #pragma unroll
    for (int r = 0; r < 8; ++r) {
        int node = base16 + rbase + r;
        float val = acc[r] + bb;
        val = val > 0.0f ? val : 0.01f * val;
        out_ffn[(size_t)node * HID + c] = val;
        atomicAdd(&pooled[batch[node] * HID + c], val);
    }
}

// ---------------- K7: z = pooled / max(cnt,1) ----------------
__global__ void k7_pool(const float* __restrict__ pooled, const float* __restrict__ cnt,
                        float* __restrict__ zout) {
    int i = blockIdx.x * 256 + threadIdx.x;
    if (i < NG * HID) {
        float c = cnt[i >> 7];
        c = c > 1.0f ? c : 1.0f;
        zout[i] = pooled[i] / c;
    }
}

// ---------------- launch ----------------
extern "C" void kernel_launch(void* const* d_in, const int* in_sizes, int n_in,
                              void* d_out, int out_size, void* d_ws, size_t ws_size,
                              hipStream_t stream) {
    const int*   x      = (const int*)d_in[0];
    const int*   esrc   = (const int*)d_in[1];
    const int*   edst   = ((const int*)d_in[1]) + EE;
    const int*   batch  = (const int*)d_in[2];
    const float* emb    = (const float*)d_in[3];
    const float* W      = (const float*)d_in[4];
    const float* att_s  = (const float*)d_in[5];
    const float* att_d  = (const float*)d_in[6];
    const float* bias   = (const float*)d_in[7];
    const float* gamma  = (const float*)d_in[8];
    const float* beta   = (const float*)d_in[9];
    const float* W2     = (const float*)d_in[10];
    const float* b2     = (const float*)d_in[11];

    float* out_ffn = (float*)d_out;               // N*128
    float* out_z   = out_ffn + (size_t)NN * HID;  // 256*128

    char* ws = (char*)d_ws;
    float*          TAB51 = (float*)(ws + 0);              //  65536 B
    float*          ASRC  = (float*)(ws + 65536);          //    512 B
    float*          ADST  = (float*)(ws + 66048);          //    512 B
    unsigned short* BPACK = (unsigned short*)(ws + 66560); //  65536 B (32B-aligned)
    float*          CNT   = (float*)(ws + 132096);         //   1024 B
    float*          POOL  = (float*)(ws + 133120);         // 131072 B
    int*            DEG   = (int*)(ws + 264192);           // 400000 B
    int*            OFFS  = (int*)(ws + 664192);           // 400004 B
    int*            CURS  = (int*)(ws + 1064448);          // 400000 B
    int*            CSRC  = (int*)(ws + 1464448);          // 2800000 B
    unsigned short* HN    = (unsigned short*)(ws + 4264448); // 51.2 MB

    // init (CNT+POOL contiguous: 33024 floats)
    kzero<<<(33024 + 255) / 256, 256, 0, stream>>>(CNT, 33024);
    k0b_init<<<(NN + 255) / 256, 256, 0, stream>>>(DEG, CNT, batch);

    // vocab-factorized precompute
    k1a_table<<<64, 256, 0, stream>>>(emb, W, TAB51);
    k1b_att<<<1, 128, 0, stream>>>(TAB51, att_s, att_d, ASRC, ADST);
    k1c_pack<<<128, 256, 0, stream>>>(W2, BPACK);

    // CSR by destination
    k2_count<<<(EE + 255) / 256, 256, 0, stream>>>(edst, DEG);
    k3_scan<<<1, 1024, 0, stream>>>(DEG, OFFS, CURS);
    k4_scatter<<<(EE + NN + 255) / 256, 256, 0, stream>>>(esrc, edst, CURS, CSRC);

    // GAT aggregate + LayerNorm -> bf16
    k5_gat<<<NN / 8, 256, 0, stream>>>(x, OFFS, CSRC, TAB51, ASRC, ADST,
                                       bias, gamma, beta, HN);

    // FFN with WMMA + pooling
    k6_ffn<<<NN / 16, 256, 0, stream>>>(HN, BPACK, b2, batch, out_ffn, POOL);

    // finalize graph means
    k7_pool<<<(NG * HID + 255) / 256, 256, 0, stream>>>(POOL, CNT, out_z);
}
